// Attention_34376918237341
// MI455X (gfx1250) — matrix operand order repaired and checked by reading, not compiled, not run on hardware
//
#include <hip/hip_runtime.h>
#include <hip/hip_bf16.h>
#include <math.h>
#include <stdint.h>

typedef __bf16 bf16_t;
typedef __attribute__((ext_vector_type(4)))  bf16_t v4bf;
typedef __attribute__((ext_vector_type(8)))  bf16_t v8bf;
typedef __attribute__((ext_vector_type(16))) bf16_t v16bf;
typedef __attribute__((ext_vector_type(8)))  float  v8f;

#define BB   4
#define TT   2048
#define HH   16
#define KVHN 4
#define HD   128
#define DD   2048
#define KVD  512
#define MM   (BB*TT)   /* 8192 */

__device__ __forceinline__ v16bf cat8(v8bf lo, v8bf hi) {
  return __builtin_shufflevector(lo, hi, 0,1,2,3,4,5,6,7,8,9,10,11,12,13,14,15);
}
__device__ __forceinline__ v8f wmma_bf16(v16bf a, v16bf b, v8f c) {
  // emits v_wmma_f32_16x16x32_bf16
  return __builtin_amdgcn_wmma_f32_16x16x32_bf16(false, a, false, b, (short)0, c, false, false);
}
__device__ __forceinline__ v8f zero8() {
  v8f z = {0.f,0.f,0.f,0.f,0.f,0.f,0.f,0.f};
  return z;
}
// CDNA5 async direct-to-LDS copy (16B per lane), tracked by ASYNCcnt.
// LDS operand = wave-relative LDS byte address = low 32 bits of the flat pointer.
__device__ __forceinline__ void async_ld16(void* lds, const void* gptr) {
  unsigned       l = (unsigned)(uintptr_t)lds;
  unsigned long long g = (unsigned long long)(uintptr_t)gptr;
  asm volatile("global_load_async_to_lds_b128 %0, %1, off" :: "v"(l), "v"(g) : "memory");
}
__device__ __forceinline__ void wait_async0() {
  asm volatile("s_wait_asynccnt 0" ::: "memory");
}

// ---------------------------------------------------------------- f32 -> bf16
__global__ __launch_bounds__(256) void cvt_f32_bf16_kernel(
    const float* __restrict__ in, bf16_t* __restrict__ out, int n)
{
  int i = (blockIdx.x * 256 + threadIdx.x) * 4;
  if (i < n) {
    float4 f = *(const float4*)(in + i);
    v4bf o;
    o[0] = (bf16_t)f.x; o[1] = (bf16_t)f.y; o[2] = (bf16_t)f.z; o[3] = (bf16_t)f.w;
    *(v4bf*)(out + i) = o;
  }
}

// ------------------------------------------------- bf16 GEMM: C = A[M,K] * B[N,K]^T
// WG tile 128x128, 8 waves (4 in M x 2 in N), wave tile 32x64.
// K-step 64, ping-pong LDS, tiles streamed with global_load_async_to_lds_b128.
template<bool OUT_F32>
__global__ __launch_bounds__(256, 2) void gemm_bf16_kernel(
    const bf16_t* __restrict__ A, const bf16_t* __restrict__ Bw,
    void* __restrict__ Cout, int M, int N, int K)
{
  __shared__ __align__(16) bf16_t As[2][128 * 64];   // 16 KB x2
  __shared__ __align__(16) bf16_t Bs[2][128 * 64];   // 16 KB x2

  const int tid  = threadIdx.x;
  const int lane = tid & 31;
  const int wv   = tid >> 5;
  const int wm   = wv & 3;      // 0..3 : 32-row band
  const int wn   = wv >> 2;     // 0..1 : 64-col band
  const int half = lane >> 4;
  const int l15  = lane & 15;
  const int blockM = blockIdx.y * 128;
  const int blockN = blockIdx.x * 128;

  v8f acc[2][4];
#pragma unroll
  for (int i = 0; i < 2; ++i)
#pragma unroll
    for (int j = 0; j < 4; ++j) acc[i][j] = zero8();

  // issue one 128x64 A tile + one 128x64 B tile: 1024 chunks each, 4+4 per thread
  auto stage = [&](int buf, int k0) {
#pragma unroll
    for (int i = 0; i < 4; ++i) {
      int c   = tid + i * 256;
      int row = c >> 3;
      int col = (c & 7) * 8;
      async_ld16(&As[buf][row * 64 + col],
                 &A[(size_t)(blockM + row) * K + k0 + col]);
      async_ld16(&Bs[buf][row * 64 + col],
                 &Bw[(size_t)(blockN + row) * K + k0 + col]);
    }
  };

  const int nk = K >> 6;
  stage(0, 0);
  for (int kt = 0; kt < nk; ++kt) {
    const int cur = kt & 1;
    wait_async0();              // my chunks for tile kt are in LDS
    __syncthreads();            // everyone's chunks are in; prev buffer fully read
    if (kt + 1 < nk) stage(cur ^ 1, (kt + 1) << 6);   // overlap with compute

#pragma unroll
    for (int ks = 0; ks < 2; ++ks) {
      v16bf a[2], b[4];
#pragma unroll
      for (int i = 0; i < 2; ++i) {   // A frag: M across lanes, K split per half-wave
        const bf16_t* r = &As[cur][(wm * 32 + i * 16 + l15) * 64 + ks * 32 + half * 8];
        a[i] = cat8(*(const v8bf*)r, *(const v8bf*)(r + 16));
      }
#pragma unroll
      for (int j = 0; j < 4; ++j) {   // B frag: N across lanes, K split per half-wave
        const bf16_t* r = &Bs[cur][(wn * 64 + j * 16 + l15) * 64 + ks * 32 + half * 8];
        b[j] = cat8(*(const v8bf*)r, *(const v8bf*)(r + 16));
      }
#pragma unroll
      for (int i = 0; i < 2; ++i)
#pragma unroll
        for (int j = 0; j < 4; ++j) acc[i][j] = wmma_bf16(a[i], b[j], acc[i][j]);
    }
    __syncthreads();            // done reading buf cur before it is restaged
  }

#pragma unroll
  for (int i = 0; i < 2; ++i)
#pragma unroll
    for (int j = 0; j < 4; ++j) {
      int n = blockN + wn * 64 + j * 16 + l15;
#pragma unroll
      for (int r = 0; r < 8; ++r) {
        int m = blockM + wm * 32 + i * 16 + half * 8 + r;
        if (OUT_F32) ((float*)Cout)[(size_t)m * N + n] = acc[i][j][r];
        else         ((bf16_t*)Cout)[(size_t)m * N + n] = (bf16_t)acc[i][j][r];
      }
    }
}

// ------------------------------- RMSNorm + RoPE (+gain & 1/sqrt(HD) on q), transpose
// in : [B, T, nh, HD] bf16   ->  out : [B, nh, T, HD] bf16.  One wave per row.
template<bool ISQ>
__global__ __launch_bounds__(256) void rope_rms_kernel(
    const bf16_t* __restrict__ in, bf16_t* __restrict__ out,
    const float* __restrict__ gain, int nh)
{
  const int lane = threadIdx.x & 31;
  const int wv   = threadIdx.x >> 5;
  const int row  = blockIdx.x * 8 + wv;     // (b*T + t)*nh + h
  const int h  = row % nh;
  const int bt = row / nh;
  const int t  = bt % TT;
  const int b  = bt / TT;

  const bf16_t* src = in + (size_t)row * HD;
  float x0 = (float)src[lane];
  float x1 = (float)src[lane + 32];
  float x2 = (float)src[lane + 64];
  float x3 = (float)src[lane + 96];

  float ss = x0*x0 + x1*x1 + x2*x2 + x3*x3;
#pragma unroll
  for (int m = 16; m >= 1; m >>= 1) ss += __shfl_xor(ss, m, 32);
  const float inv = rsqrtf(ss * (1.0f / HD) + 1.1920928955078125e-07f);
  x0 *= inv; x1 *= inv; x2 *= inv; x3 *= inv;

  const float LN = 9.210340371976184f;  // ln(10000)
  float th0 = (float)t * __expf(-(float)lane        * (LN / 64.0f));
  float th1 = (float)t * __expf(-(float)(lane + 32) * (LN / 64.0f));
  float s0, c0, s1, c1;
  __sincosf(th0, &s0, &c0);
  __sincosf(th1, &s1, &c1);

  float g = 1.0f;
  if (ISQ) g = gain[h] * 0.08838834764831845f;  // fold 1/sqrt(HD) into q

  float o0 = (x0 * c0 - x2 * s0) * g;
  float o1 = (x1 * c1 - x3 * s1) * g;
  float o2 = (x2 * c0 + x0 * s0) * g;
  float o3 = (x3 * c1 + x1 * s1) * g;

  bf16_t* dst = out + (((size_t)(b * nh + h)) * TT + t) * HD;
  dst[lane]      = (bf16_t)o0;
  dst[lane + 32] = (bf16_t)o1;
  dst[lane + 64] = (bf16_t)o2;
  dst[lane + 96] = (bf16_t)o3;
}

// ------------------------------------------- v transpose: [B,T,KVH,HD] -> [B,KVH,T,HD]
__global__ __launch_bounds__(256) void vtrans_kernel(
    const bf16_t* __restrict__ in, bf16_t* __restrict__ out)
{
  size_t idx = ((size_t)blockIdx.x * 256 + threadIdx.x) * 8;
  int d0 = (int)(idx & 127);
  int hh = (int)((idx >> 7) & 3);
  int t  = (int)((idx >> 9) & 2047);
  int b  = (int)(idx >> 20);
  v8bf val = *(const v8bf*)(in + idx);
  *(v8bf*)(out + (((size_t)(b * KVHN + hh)) * TT + t) * HD + d0) = val;
}

// ------------------------------------------------------------- flash attention
// One WG per (b, h, 128-row q tile). 8 waves x 16 q-rows. KV tiles of 64 keys.
// q/k/v: [B, H|KVH, T, HD] bf16.  o: [B, T, D] bf16 (ready for proj GEMM).
__global__ __launch_bounds__(256, 1) void flash_attn_kernel(
    const bf16_t* __restrict__ q, const bf16_t* __restrict__ k,
    const bf16_t* __restrict__ v, bf16_t* __restrict__ o)
{
  __shared__ __align__(16) bf16_t Ks[64 * 128];    // [key][d]
  __shared__ __align__(16) bf16_t Vt[128 * 64];    // [d][key]
  __shared__ __align__(16) bf16_t Ps[8][16 * 64];  // per-wave P relayout

  const int tid  = threadIdx.x;
  const int lane = tid & 31;
  const int wv   = tid >> 5;
  const int half = lane >> 4;
  const int l15  = lane & 15;

  const int i   = blockIdx.x & 15;        // q tile index (128 rows)
  const int bh  = blockIdx.x >> 4;
  const int h   = bh & (HH - 1);
  const int b   = bh >> 4;
  const int kvh = h >> 2;                 // GQA group

  const bf16_t* Qg = q + (((size_t)(b * HH + h)) * TT + i * 128) * HD;
  const bf16_t* Kg = k + ((size_t)(b * KVHN + kvh)) * TT * HD;
  const bf16_t* Vg = v + ((size_t)(b * KVHN + kvh)) * TT * HD;

  // Q fragments straight from global (per-wave 16 rows x full HD, pre-scaled)
  v16bf aq[4];
#pragma unroll
  for (int kc = 0; kc < 4; ++kc) {
    const bf16_t* r = Qg + (wv * 16 + l15) * HD + kc * 32 + half * 8;
    aq[kc] = cat8(*(const v8bf*)r, *(const v8bf*)(r + 16));
  }

  v8f oacc[8];
#pragma unroll
  for (int dt = 0; dt < 8; ++dt) oacc[dt] = zero8();
  float mrow[8], lrow[8];
#pragma unroll
  for (int r = 0; r < 8; ++r) { mrow[r] = -INFINITY; lrow[r] = 0.0f; }

  const int jmax = 2 * i + 1;
  for (int j = 0; j <= jmax; ++j) {
    __syncthreads();
#pragma unroll
    for (int c0 = 0; c0 < 4; ++c0) {     // stage K (async, direct-to-LDS) + V (transposed)
      int c   = tid + c0 * 256;
      int row = c >> 4;
      int col = (c & 15) * 8;
      async_ld16(&Ks[row * 128 + col], &Kg[(size_t)(j * 64 + row) * HD + col]);
      v8bf vv = *(const v8bf*)&Vg[(size_t)(j * 64 + row) * HD + col];
#pragma unroll
      for (int e = 0; e < 8; ++e) Vt[(col + e) * 64 + row] = vv[e];
    }
    wait_async0();
    __syncthreads();

    // prefetch next KV tile into cache while we compute on this one
    if (j < jmax) {
      const size_t poff = (size_t)((j + 1) * 64 + (tid >> 2)) * HD + (tid & 3) * 32;
      __builtin_prefetch(&Kg[poff], 0, 0);   // global_prefetch_b8
      __builtin_prefetch(&Vg[poff], 0, 0);
    }

    // S = Q K^T   (wave's 16 rows x 64 keys)
    v8f sacc[4];
#pragma unroll
    for (int nt = 0; nt < 4; ++nt) sacc[nt] = zero8();
#pragma unroll
    for (int nt = 0; nt < 4; ++nt)
#pragma unroll
      for (int kc = 0; kc < 4; ++kc) {
        const bf16_t* r = &Ks[(nt * 16 + l15) * 128 + kc * 32 + half * 8];
        v16bf bk = cat8(*(const v8bf*)r, *(const v8bf*)(r + 16));
        sacc[nt] = wmma_bf16(aq[kc], bk, sacc[nt]);
      }

    // causal mask (only tiles intersecting the diagonal need it)
    if (j >= 2 * i) {
#pragma unroll
      for (int nt = 0; nt < 4; ++nt)
#pragma unroll
        for (int r = 0; r < 8; ++r) {
          int kg = j * 64 + nt * 16 + l15;
          int qr = i * 128 + wv * 16 + half * 8 + r;
          if (kg > qr) sacc[nt][r] = -INFINITY;
        }
    }

    // online softmax
    float mnew[8], alpha[8], rs[8];
#pragma unroll
    for (int r = 0; r < 8; ++r) {
      float mx = fmaxf(fmaxf(sacc[0][r], sacc[1][r]),
                       fmaxf(sacc[2][r], sacc[3][r]));
#pragma unroll
      for (int msk = 8; msk >= 1; msk >>= 1) mx = fmaxf(mx, __shfl_xor(mx, msk, 32));
      mnew[r]  = fmaxf(mrow[r], mx);
      alpha[r] = __expf(mrow[r] - mnew[r]);
      mrow[r]  = mnew[r];
      rs[r]    = 0.0f;
    }
#pragma unroll
    for (int nt = 0; nt < 4; ++nt)
#pragma unroll
      for (int r = 0; r < 8; ++r) {
        float p = __expf(sacc[nt][r] - mnew[r]);
        sacc[nt][r] = p;
        rs[r] += p;
      }
#pragma unroll
    for (int r = 0; r < 8; ++r) {
#pragma unroll
      for (int msk = 8; msk >= 1; msk >>= 1) rs[r] += __shfl_xor(rs[r], msk, 32);
      lrow[r] = lrow[r] * alpha[r] + rs[r];
    }
#pragma unroll
    for (int dt = 0; dt < 8; ++dt)
#pragma unroll
      for (int r = 0; r < 8; ++r) oacc[dt][r] *= alpha[r];

    // relayout P (C-layout -> A-layout) through per-wave LDS
    bf16_t* P = &Ps[wv][0];
#pragma unroll
    for (int nt = 0; nt < 4; ++nt)
#pragma unroll
      for (int r = 0; r < 8; ++r)
        P[(half * 8 + r) * 64 + nt * 16 + l15] = (bf16_t)sacc[nt][r];
    asm volatile("s_wait_dscnt 0" ::: "memory");  // wave-local LDS RAW fence

    // O += P V
    v16bf ap[2];
#pragma unroll
    for (int kc = 0; kc < 2; ++kc) {
      const bf16_t* r = &P[l15 * 64 + kc * 32 + half * 8];
      ap[kc] = cat8(*(const v8bf*)r, *(const v8bf*)(r + 16));
    }
#pragma unroll
    for (int dt = 0; dt < 8; ++dt)
#pragma unroll
      for (int kc = 0; kc < 2; ++kc) {
        const bf16_t* r = &Vt[(dt * 16 + l15) * 64 + kc * 32 + half * 8];
        v16bf bv = cat8(*(const v8bf*)r, *(const v8bf*)(r + 16));
        oacc[dt] = wmma_bf16(ap[kc], bv, oacc[dt]);
      }
  }

  // normalize and store [B, T, D]
  float linv[8];
#pragma unroll
  for (int r = 0; r < 8; ++r) linv[r] = 1.0f / lrow[r];
#pragma unroll
  for (int dt = 0; dt < 8; ++dt)
#pragma unroll
    for (int r = 0; r < 8; ++r) {
      int t   = i * 128 + wv * 16 + half * 8 + r;
      int col = h * HD + dt * 16 + l15;
      o[((size_t)(b * TT + t)) * DD + col] = (bf16_t)(oacc[dt][r] * linv[r]);
    }
}

// ---------------------------------------------------------------------- launch
extern "C" void kernel_launch(void* const* d_in, const int* in_sizes, int n_in,
                              void* d_out, int out_size, void* d_ws, size_t ws_size,
                              hipStream_t stream)
{
  const float* x  = (const float*)d_in[0];
  const float* Wq = (const float*)d_in[1];
  const float* Wk = (const float*)d_in[2];
  const float* Wv = (const float*)d_in[3];
  const float* Wp = (const float*)d_in[4];
  const float* qg = (const float*)d_in[5];

  char* w = (char*)d_ws;
  bf16_t* xb  = (bf16_t*)w; w += (size_t)MM * DD * 2;
  bf16_t* Wqb = (bf16_t*)w; w += (size_t)DD * DD * 2;
  bf16_t* Wkb = (bf16_t*)w; w += (size_t)KVD * DD * 2;
  bf16_t* Wvb = (bf16_t*)w; w += (size_t)KVD * DD * 2;
  bf16_t* Wpb = (bf16_t*)w; w += (size_t)DD * DD * 2;
  bf16_t* qb  = (bf16_t*)w; w += (size_t)MM * DD * 2;
  bf16_t* kb  = (bf16_t*)w; w += (size_t)MM * KVD * 2;
  bf16_t* vb  = (bf16_t*)w; w += (size_t)MM * KVD * 2;
  bf16_t* qT  = (bf16_t*)w; w += (size_t)MM * DD * 2;
  bf16_t* kT  = (bf16_t*)w; w += (size_t)MM * KVD * 2;
  bf16_t* vT  = (bf16_t*)w; w += (size_t)MM * KVD * 2;
  bf16_t* ob  = xb;  // x dead after projections; reuse for attention output

  // 1) convert to bf16
  cvt_f32_bf16_kernel<<<(MM * DD) / 1024, 256, 0, stream>>>(x,  xb,  MM * DD);
  cvt_f32_bf16_kernel<<<(DD * DD) / 1024, 256, 0, stream>>>(Wq, Wqb, DD * DD);
  cvt_f32_bf16_kernel<<<(KVD * DD) / 1024, 256, 0, stream>>>(Wk, Wkb, KVD * DD);
  cvt_f32_bf16_kernel<<<(KVD * DD) / 1024, 256, 0, stream>>>(Wv, Wvb, KVD * DD);
  cvt_f32_bf16_kernel<<<(DD * DD) / 1024, 256, 0, stream>>>(Wp, Wpb, DD * DD);

  // 2) QKV projections (bf16 WMMA, f32 accum, async double-buffered staging)
  gemm_bf16_kernel<false><<<dim3(DD / 128,  MM / 128), 256, 0, stream>>>(xb, Wqb, qb, MM, DD,  DD);
  gemm_bf16_kernel<false><<<dim3(KVD / 128, MM / 128), 256, 0, stream>>>(xb, Wkb, kb, MM, KVD, DD);
  gemm_bf16_kernel<false><<<dim3(KVD / 128, MM / 128), 256, 0, stream>>>(xb, Wvb, vb, MM, KVD, DD);

  // 3) RMSNorm + RoPE (+gain/scale on q) with head-major transpose; v transpose
  rope_rms_kernel<true ><<<(MM * HH)   / 8, 256, 0, stream>>>(qb, qT, qg, HH);
  rope_rms_kernel<false><<<(MM * KVHN) / 8, 256, 0, stream>>>(kb, kT, qg, KVHN);
  vtrans_kernel<<<(MM * KVD / 8) / 256, 256, 0, stream>>>(vb, vT);

  // 4) causal flash attention (WMMA QK^T and PV, online softmax)
  flash_attn_kernel<<<BB * HH * (TT / 128), 256, 0, stream>>>(qT, kT, vT, ob);

  // 5) output projection, f32 result straight to d_out
  gemm_bf16_kernel<true><<<dim3(DD / 128, MM / 128), 256, 0, stream>>>(ob, Wpb, d_out, MM, DD, DD);
}
